// NPUOptimizedRK4_82257213653103
// MI455X (gfx1250) — matrix operand order repaired and checked by reading, not compiled, not run on hardware
//
#include <hip/hip_runtime.h>

// Neural-ODE RK4 on MI455X (gfx1250): all matmuls through v_wmma_f32_16x16x32_f16.
// A operands are row-major [M][K]; B operands are pre-transposed to [N][K] so every
// lane's WMMA fragment is two contiguous 16B global_load_b128s (per CDNA5 ISA
// 16-bit A 16x32 / B 32x16 layouts). Weights+activations fit in L2 (192MB), so no
// LDS staging; the kernel is matrix-core bound. tanh is a branch-free
// v_exp_f32/v_rcp_f32 sequence instead of branchy libm.

#define BATCH  4096
#define DIM    512
#define HID    2048
#define KPAD   544          // (DIM+1)=513 padded up to a multiple of 32
#define NSTEPS 20

typedef __attribute__((ext_vector_type(16))) _Float16 v16h;
typedef __attribute__((ext_vector_type(8)))  float    v8f;
typedef __attribute__((ext_vector_type(4)))  float    f4v;   // 16-byte load unit

union HFrag { v16h v; f4v q[2]; };

// Branch-free tanh: tanh(x) = (e-1)/(e+1), e = 2^(2*log2(e)*x).
// Clamp |x|<=15 so e <= 2^43.3 stays finite (no inf/inf), no EXEC divergence.
__device__ __forceinline__ float fast_tanh(float x) {
  float cx = fminf(15.0f, fmaxf(-15.0f, x));
  float e  = __builtin_amdgcn_exp2f(cx * 2.8853900817779268f);  // 2/ln(2)
  return (e - 1.0f) * __builtin_amdgcn_rcpf(e + 1.0f);
}

// ---------------- weight convert + transpose (run once per launch) ----------------
__global__ void conv_w1(const float* __restrict__ W1, _Float16* __restrict__ W1t) {
  int idx = blockIdx.x * blockDim.x + threadIdx.x;
  if (idx >= HID * KPAD) return;
  int n = idx / KPAD, k = idx - n * KPAD;
  float v = (k < DIM + 1) ? W1[(size_t)k * HID + n] : 0.0f;
  W1t[idx] = (_Float16)v;
}

__global__ void conv_w2(const float* __restrict__ W2, _Float16* __restrict__ W2t) {
  int idx = blockIdx.x * blockDim.x + threadIdx.x;
  if (idx >= DIM * HID) return;
  int n = idx / HID, k = idx - n * HID;
  W2t[idx] = (_Float16)W2[(size_t)k * DIM + n];
}

// ---------------- build stage input: Zb = f16([z + coef*h*k , t , 0-pad]) ----------
__global__ void prep_stage(const float* __restrict__ z, const float* __restrict__ kk,
                           const float* __restrict__ t0p, const float* __restrict__ t1p,
                           float coefFrac, float timeFrac, _Float16* __restrict__ Zb) {
  int idx = blockIdx.x * blockDim.x + threadIdx.x;
  if (idx >= BATCH * KPAD) return;
  int row = idx / KPAD, col = idx - row * KPAD;
  float h = (t1p[0] - t0p[0]) * (1.0f / NSTEPS);
  float v;
  if (col < DIM)       v = z[(size_t)row * DIM + col] + coefFrac * h * kk[(size_t)row * DIM + col];
  else if (col == DIM) v = t0p[0] + timeFrac * h;
  else                 v = 0.0f;
  Zb[idx] = (_Float16)v;
}

// ---------------- RK4 accumulate: zdst = (init? zsrc : zdst) + wFrac*h*k ----------
__global__ void rk4_accum(const float* __restrict__ zsrc, const float* __restrict__ kk,
                          const float* __restrict__ t0p, const float* __restrict__ t1p,
                          float wFrac, int init, float* __restrict__ zdst) {
  int idx = blockIdx.x * blockDim.x + threadIdx.x;
  if (idx >= BATCH * DIM) return;
  float h = (t1p[0] - t0p[0]) * (1.0f / NSTEPS);
  float base = init ? zsrc[idx] : zdst[idx];
  zdst[idx] = base + wFrac * h * kk[idx];
}

// ---------------- WMMA GEMM: C[M][ldc] = epi(A[M][K] * Bt[N][K]^T + bias) ----------
// Block = 128 threads (4 waves), block tile 128x128, wave tile 64x64 (4x4 WMMA tiles).
template <bool TANH_F16_OUT>
__global__ __launch_bounds__(128) void wmma_gemm(
    const _Float16* __restrict__ A, const _Float16* __restrict__ Bt,
    const float* __restrict__ bias, void* __restrict__ outv,
    int K, int ldc) {
  const int lane = threadIdx.x & 31;
  const int wave = threadIdx.x >> 5;        // 0..3
  const int half = lane >> 4;               // 0: lanes 0-15, 1: lanes 16-31
  const int l16  = lane & 15;
  const int m0 = blockIdx.y * 128 + (wave >> 1) * 64;
  const int n0 = blockIdx.x * 128 + (wave & 1) * 64;

  v8f c[4][4] = {};

#pragma unroll 1
  for (int kb = 0; kb < K; kb += 32) {
    HFrag a[4], b[4];
    // A fragment (16-bit A 16x32): lanes 0-15 hold K=[0..7],[16..23] of row M=l16;
    // lanes 16-31 hold K=[8..15],[24..31] of the same rows.
#pragma unroll
    for (int i = 0; i < 4; ++i) {
      const _Float16* ap = A + (size_t)(m0 + i * 16 + l16) * K + kb + half * 8;
      a[i].q[0] = *(const f4v*)ap;
      a[i].q[1] = *(const f4v*)(ap + 16);
    }
    // B fragment (16-bit B 32x16): lanes 0-15 hold K=[0..7],[16..23] of column N=l16;
    // lanes 16-31 hold K=[8..15],[24..31]. Bt is [N][K] so each half is 16 contiguous bytes.
#pragma unroll
    for (int j = 0; j < 4; ++j) {
      const _Float16* bp = Bt + (size_t)(n0 + j * 16 + l16) * K + kb + half * 8;
      b[j].q[0] = *(const f4v*)bp;
      b[j].q[1] = *(const f4v*)(bp + 16);
    }
#pragma unroll
    for (int i = 0; i < 4; ++i)
#pragma unroll
      for (int j = 0; j < 4; ++j)
        c[i][j] = __builtin_amdgcn_wmma_f32_16x16x32_f16(
            false, a[i].v, false, b[j].v, (short)0, c[i][j], false, false);
  }

  // C/D layout (32-bit 16x16): lane half 0 -> rows M=0..7, half 1 -> rows M=8..15,
  // column N = l16; VGPR r holds row (half*8 + r).
#pragma unroll
  for (int i = 0; i < 4; ++i) {
#pragma unroll
    for (int j = 0; j < 4; ++j) {
      const int row0 = m0 + i * 16 + half * 8;
      const int col  = n0 + j * 16 + l16;
      const float bv = bias[col];
#pragma unroll
      for (int r = 0; r < 8; ++r) {
        float v = c[i][j][r] + bv;
        if (TANH_F16_OUT) {
          ((_Float16*)outv)[(size_t)(row0 + r) * ldc + col] = (_Float16)fast_tanh(v);
        } else {
          ((float*)outv)[(size_t)(row0 + r) * ldc + col] = v;
        }
      }
    }
  }
}

// ------------------------------------ host -----------------------------------------
extern "C" void kernel_launch(void* const* d_in, const int* in_sizes, int n_in,
                              void* d_out, int out_size, void* d_ws, size_t ws_size,
                              hipStream_t stream) {
  (void)in_sizes; (void)n_in; (void)out_size; (void)ws_size;
  const float* z0 = (const float*)d_in[0];
  const float* W1 = (const float*)d_in[1];
  const float* b1 = (const float*)d_in[2];
  const float* W2 = (const float*)d_in[3];
  const float* b2 = (const float*)d_in[4];
  const float* t0 = (const float*)d_in[5];
  const float* t1 = (const float*)d_in[6];
  float* out = (float*)d_out;

  // Workspace carve-up (all offsets 256B aligned); total ~50.7 MB.
  char* ws = (char*)d_ws;
  _Float16* W1t = (_Float16*)(ws + 0);          // HID*KPAD*2  = 2,228,224
  _Float16* W2t = (_Float16*)(ws + 2228224);    // DIM*HID*2   = 2,097,152
  _Float16* Zb  = (_Float16*)(ws + 4325376);    // BATCH*KPAD*2= 4,456,448
  _Float16* Hbuf= (_Float16*)(ws + 8781824);    // BATCH*HID*2 = 16,777,216
  float*    kbuf= (float*)   (ws + 25559040);   // BATCH*DIM*4 = 8,388,608
  float*    zA  = (float*)   (ws + 33947648);   // BATCH*DIM*4
  float*    zB  = (float*)   (ws + 42336256);   // BATCH*DIM*4

  conv_w1<<<(HID * KPAD + 255) / 256, 256, 0, stream>>>(W1, W1t);
  conv_w2<<<(DIM * HID + 255) / 256, 256, 0, stream>>>(W2, W2t);

  const dim3 g1(HID / 128, BATCH / 128);   // 16 x 32 blocks
  const dim3 g2(DIM / 128, BATCH / 128);   //  4 x 32 blocks
  const int pgrid = (BATCH * KPAD + 255) / 256;
  const int agrid = (BATCH * DIM + 255) / 256;

  const float cf[4] = {0.0f, 0.5f, 0.5f, 1.0f};             // z-stage coefficient (×h)
  const float tf[4] = {0.0f, 0.5f, 0.5f, 1.0f};             // time fraction within step
  const float wf[4] = {1.f/6.f, 1.f/3.f, 1.f/3.f, 1.f/6.f}; // RK4 weights (×h)

  const float* zcur = z0;
  for (int s = 0; s < NSTEPS; ++s) {
    float* zdst = (s == NSTEPS - 1) ? out : ((s & 1) ? zB : zA);
    for (int st = 0; st < 4; ++st) {
      const float* kptr = (st == 0) ? zcur : kbuf;  // coef=0 on stage 1, any finite ptr
      prep_stage<<<pgrid, 256, 0, stream>>>(zcur, kptr, t0, t1,
                                            cf[st], (float)s + tf[st], Zb);
      wmma_gemm<true ><<<g1, 128, 0, stream>>>(Zb,   W1t, b1, (void*)Hbuf, KPAD, HID);
      wmma_gemm<false><<<g2, 128, 0, stream>>>(Hbuf, W2t, b2, (void*)kbuf, HID,  DIM);
      rk4_accum<<<agrid, 256, 0, stream>>>(zcur, kbuf, t0, t1,
                                           wf[st], st == 0 ? 1 : 0, zdst);
    }
    zcur = zdst;
  }
}
